// Tacotron_18700287607106
// MI455X (gfx1250) — compile-verified, matrix operand order checked by reading, not run on hardware
//
#include <hip/hip_runtime.h>
#include <hip/hip_bf16.h>

// ---------------------------------------------------------------------------
// Tacotron decoder for MI455X (gfx1250, wave32, WMMA bf16 16x16x32)
//   B=64, T_DEC=200, T_ENC=512, D=256, IN=400 (padded to 416 for K%32==0)
// ---------------------------------------------------------------------------

#define BATCH   64
#define TDEC    200
#define TENC    512
#define DMODEL  256
#define INRAW   400
#define INPAD   416
#define HID     256
#define GATES   768
#define MROWS   (BATCH * TDEC)      // 12800
#define EROWS   (BATCH * TENC)      // 32768
#define NWG     16
#define RBLK    512

typedef __attribute__((ext_vector_type(16))) __bf16        bf16x16;
typedef __attribute__((ext_vector_type(8)))  float         f32x8;
typedef __attribute__((ext_vector_type(8)))  unsigned short u16x8;

// float -> bf16 (round to nearest even), pure integer math (no __bf16 arith)
static __device__ __forceinline__ unsigned short f2bf(float f) {
    union { float f; unsigned u; } x; x.f = f;
    unsigned r = x.u + 0x7FFFu + ((x.u >> 16) & 1u);
    return (unsigned short)(r >> 16);
}

static __device__ __forceinline__ float sigm(float x) {
    return 1.f / (1.f + __expf(-x));
}
static __device__ __forceinline__ float tanh_f(float x) {
    x = fminf(fmaxf(x, -15.f), 15.f);
    float e = __expf(2.f * x);
    return (e - 1.f) / (e + 1.f);
}

// Load one 16x32 bf16 WMMA fragment (A or Bt), row-major source, stride in
// elements.  Per the CDNA5 ISA 16-bit A layout, lane l (m = l&15, hi = l>>4)
// holds k = hi*8 + e (e<8) and k = 16 + hi*8 + (e-8) (e>=8): two contiguous
// 16-byte loads per lane.
static __device__ __forceinline__ bf16x16 frag_ld(const unsigned short* base,
                                                  long stride, int row0, int k0) {
    int lane = threadIdx.x & 31;
    const unsigned short* rp =
        base + (long)(row0 + (lane & 15)) * stride + k0 + ((lane >> 4) << 3);
    union { u16x8 u[2]; bf16x16 b; } f;
    f.u[0] = *(const u16x8*)rp;
    f.u[1] = *(const u16x8*)(rp + 16);
    return f.b;
}

// acc += A[arow0:+16, ka0:+nk*32] * Bt[brow0:+16, kb0:+nk*32]^T  (bf16 WMMA)
static __device__ __forceinline__ f32x8 mm(f32x8 acc,
                                           const unsigned short* A, long lda, int arow0, int ka0,
                                           const unsigned short* Bt, long ldb, int brow0, int kb0,
                                           int nk) {
#pragma unroll
    for (int i = 0; i < nk; ++i) {
        bf16x16 a = frag_ld(A, lda, arow0, ka0 + i * 32);
        bf16x16 b = frag_ld(Bt, ldb, brow0, kb0 + i * 32);
        acc = __builtin_amdgcn_wmma_f32_16x16x32_bf16(false, a, false, b,
                                                      (short)0, acc, false, false);
    }
    return acc;
}

// ---------------------------------------------------------------------------
// Prep kernels
// ---------------------------------------------------------------------------

// dec_inputs = [zeros, mel[:, :-1]] -> bf16, K padded 400->416
__global__ void k_build_apre(const float* __restrict__ mel, unsigned short* __restrict__ apre) {
    long n = (long)MROWS * INPAD;
    long i = (long)blockIdx.x * blockDim.x + threadIdx.x;
    if (i >= n) return;
    long row = i / INPAD; int k = (int)(i % INPAD);
    int b = (int)(row / TDEC), t = (int)(row % TDEC);
    float v = 0.f;
    if (k < INRAW && t > 0) v = mel[((long)b * TDEC + (t - 1)) * INRAW + k];
    apre[i] = f2bf(v);
}

__global__ void k_f2bf(const float* __restrict__ src, unsigned short* __restrict__ dst, long n) {
    long i = (long)blockIdx.x * blockDim.x + threadIdx.x;
    if (i < n) dst[i] = f2bf(src[i]);
}

// w[K][N] fp32 -> wt[N][Kpad] bf16 (transposed, zero-padded)
__global__ void k_wtrans(const float* __restrict__ w, unsigned short* __restrict__ wt,
                         int K, int N, int Kpad) {
    long n = (long)N * Kpad;
    long i = (long)blockIdx.x * blockDim.x + threadIdx.x;
    if (i >= n) return;
    int nn = (int)(i / Kpad), k = (int)(i % Kpad);
    wt[i] = (k < K) ? f2bf(w[(long)k * N + nn]) : (unsigned short)0;
}

// ---------------------------------------------------------------------------
// Generic WMMA GEMM:  C[M,N] = act(A_bf[M,Kp] @ Bt_bf[N,Kp]^T + bias)
// One 16x16 tile per wave, 8 waves / WG.
// ---------------------------------------------------------------------------
__global__ void k_gemm(const unsigned short* __restrict__ A, long lda,
                       const unsigned short* __restrict__ Bt, long ldb,
                       const float* __restrict__ bias,
                       float* __restrict__ Cf, unsigned short* __restrict__ Cb, long ldc,
                       int Mt, int Nt, int Kp, int relu) {
    int tile = blockIdx.x * 8 + (threadIdx.x >> 5);
    if (tile >= Mt * Nt) return;
    int mt = tile / Nt, nt = tile % Nt;
    f32x8 acc = {};
    acc = mm(acc, A, lda, mt * 16, 0, Bt, ldb, nt * 16, 0, Kp / 32);
    int lane = threadIdx.x & 31, hi = lane >> 4, ln = lane & 15;
    int col = nt * 16 + ln;
    float bv = bias ? bias[col] : 0.f;
#pragma unroll
    for (int r = 0; r < 8; ++r) {
        long row = (long)mt * 16 + r + 8 * hi;
        float v = acc[r] + bv;
        if (relu) v = fmaxf(v, 0.f);
        if (Cf) Cf[row * ldc + col] = v;
        if (Cb) Cb[row * ldc + col] = f2bf(v);
    }
}

// ---------------------------------------------------------------------------
// Persistent recurrent kernel (16 WGs x 512 threads)
// ---------------------------------------------------------------------------
struct RecArgs {
    // bf16 transposed weights
    const unsigned short *attiht, *atthht, *qwt, *projwt;
    const unsigned short *d1iht, *d1hht, *d2iht, *d2hht, *melwt;
    // fp32 params (from d_in)
    const float *att_bih, *att_bhh, *proj_b;
    const float *d1_bih, *d1_bhh, *d2_bih, *d2_bhh, *mel_b, *att_v;
    const float *enc;                 // [B, TENC, D] fp32
    const float *PM;                  // proc_mem [B*TENC, D] fp32
    const unsigned short *XF;         // prenet out [B*TDEC, 128] bf16
    // states (ping-pong where a stage's GEMM-readers overlap combine-writers)
    float *att_h[2]; unsigned short *att_hb[2];
    float *h1[2];    unsigned short *h1b[2];
    float *h2[2];    unsigned short *h2b[2];
    float *ctx;      unsigned short *ctxb;
    float *q;
    float *dA, *dB, *dC;
    unsigned short *dAb, *dBb, *dCb;
    float *e;                         // [B, TENC]
    float *out_mel;                   // d_out
    float *out_align;                 // d_out + B*T*400
    unsigned *bar;                    // {count, generation}
};

static __device__ __forceinline__ void gsync(unsigned* bar) {
    __threadfence();
    __syncthreads();
    if (threadIdx.x == 0) {
        // CDNA5 cluster split-barrier: HW-fast when launched as one 16-WG
        // cluster, architectural NOP (ClusterID==0) otherwise.
        __builtin_amdgcn_s_cluster_barrier();
        unsigned gen = atomicAdd(&bar[1], 0u);
        if (atomicAdd(&bar[0], 1u) == NWG - 1u) {
            atomicExch(&bar[0], 0u);
            __threadfence();
            atomicAdd(&bar[1], 1u);
        } else {
            while (atomicAdd(&bar[1], 0u) == gen) __builtin_amdgcn_s_sleep(1);
        }
    }
    __syncthreads();
    __threadfence();
}

// Decoder GRU stage: this WG owns hidden cols [wg*16, wg*16+16).
// gi = d @ w_ih (K=256), gh = h_prev @ w_hh (K=256); combine + residual.
static __device__ __forceinline__ void dec_gru_stage(
    int wg, int wave, int hi, int ln, int tid,
    const unsigned short* din_b, const float* din_f,
    const unsigned short* hprev_b, const float* hprev_f,
    const unsigned short* iht, const unsigned short* hht,
    const float* bih, const float* bhh,
    float* h_f, unsigned short* h_b,
    float* dn_f, unsigned short* dn_b,
    float (*gi_s)[64][16], float (*gh_s)[64][16]) {
    if (wave < 12) {
        int gate = wave >> 2, mt = wave & 3;
        int gcol = gate * 256 + wg * 16;
        f32x8 gi = {}, gh = {};
        gi = mm(gi, din_b, 256, mt * 16, 0, iht, 256, gcol, 0, 8);
        gh = mm(gh, hprev_b, 256, mt * 16, 0, hht, 256, gcol, 0, 8);
        int col = gcol + ln;
        float bi = bih[col], bh = bhh[col];
#pragma unroll
        for (int r = 0; r < 8; ++r) {
            int bm = mt * 16 + r + 8 * hi;
            gi_s[gate][bm][ln] = gi[r] + bi;
            gh_s[gate][bm][ln] = gh[r] + bh;
        }
    }
    __syncthreads();
    for (int i = tid; i < 1024; i += RBLK) {
        int b = i >> 4, j = i & 15, jg = wg * 16 + j;
        float r = sigm(gi_s[0][b][j] + gh_s[0][b][j]);
        float z = sigm(gi_s[1][b][j] + gh_s[1][b][j]);
        float n = tanh_f(gi_s[2][b][j] + r * gh_s[2][b][j]);
        float hN = (1.f - z) * n + z * hprev_f[b * 256 + jg];
        h_f[b * 256 + jg] = hN;
        h_b[b * 256 + jg] = f2bf(hN);
        float dn = hN + din_f[b * 256 + jg];
        dn_f[b * 256 + jg] = dn;
        dn_b[b * 256 + jg] = f2bf(dn);
    }
}

__global__ void __launch_bounds__(RBLK) k_rec(RecArgs A) {
    const int wg = blockIdx.x;
    const int tid = threadIdx.x;
    const int wave = tid >> 5, lane = tid & 31, hi = lane >> 4, ln = lane & 15;

    __shared__ float gi_s[3][64][16];
    __shared__ float gh_s[3][64][16];
    __shared__ float aLds[4][512];
    __shared__ float redA[16], redB[16];

    for (int t = 0; t < TDEC; ++t) {
        const int cur = t & 1, prv = cur ^ 1;
        if (t + 1 < TDEC)
            __builtin_prefetch(A.XF + (long)(t + 1) * 128, 0, 1);

        // ---- Stage 1: attention GRU.  gi = [x_t, ctx] @ w_ih, gh = h @ w_hh.
        if (wave < 12) {
            int gate = wave >> 2, mt = wave & 3;
            int gcol = gate * 256 + wg * 16;
            f32x8 gi = {}, gh = {};
            // x part: XF rows are b*TDEC+t -> base + t*128, stride TDEC*128
            gi = mm(gi, A.XF + (long)t * 128, (long)TDEC * 128, mt * 16, 0,
                    A.attiht, 384, gcol, 0, 4);
            gi = mm(gi, A.ctxb, 256, mt * 16, 0, A.attiht, 384, gcol, 128, 8);
            gh = mm(gh, A.att_hb[prv], 256, mt * 16, 0, A.atthht, 256, gcol, 0, 8);
            int col = gcol + ln;
            float bi = A.att_bih[col], bh = A.att_bhh[col];
#pragma unroll
            for (int r = 0; r < 8; ++r) {
                int bm = mt * 16 + r + 8 * hi;
                gi_s[gate][bm][ln] = gi[r] + bi;
                gh_s[gate][bm][ln] = gh[r] + bh;
            }
        }
        __syncthreads();
        for (int i = tid; i < 1024; i += RBLK) {
            int b = i >> 4, j = i & 15, jg = wg * 16 + j;
            float r = sigm(gi_s[0][b][j] + gh_s[0][b][j]);
            float z = sigm(gi_s[1][b][j] + gh_s[1][b][j]);
            float n = tanh_f(gi_s[2][b][j] + r * gh_s[2][b][j]);
            float hN = (1.f - z) * n + z * A.att_h[prv][b * 256 + jg];
            A.att_h[cur][b * 256 + jg] = hN;
            A.att_hb[cur][b * 256 + jg] = f2bf(hN);
        }
        gsync(A.bar);

        // ---- Stage 2: q = att_h @ query_w   (this WG: cols wg*16..+16)
        if (wave < 4) {
            f32x8 acc = {};
            acc = mm(acc, A.att_hb[cur], 256, wave * 16, 0, A.qwt, 256, wg * 16, 0, 8);
#pragma unroll
            for (int r = 0; r < 8; ++r) {
                int bm = wave * 16 + r + 8 * hi;
                A.q[bm * 256 + wg * 16 + ln] = acc[r];
            }
        }
        gsync(A.bar);

        // ---- Stage 3: e[b,tp] = v . tanh(PM[b,tp,:] + q[b,:])
        for (int rep = 0; rep < 4; ++rep) {
            int item = tid + rep * RBLK;               // 0..2047
            int b = item >> 5;
            int tp = wg * 32 + (item & 31);
            const float4* pm4 = (const float4*)(A.PM + ((long)b * TENC + tp) * 256);
            const float4* q4  = (const float4*)(A.q + b * 256);
            const float4* v4  = (const float4*)A.att_v;
            float s = 0.f;
#pragma unroll 4
            for (int k = 0; k < 64; ++k) {
                float4 p = pm4[k], qq = q4[k], vv = v4[k];
                s += vv.x * tanh_f(p.x + qq.x) + vv.y * tanh_f(p.y + qq.y) +
                     vv.z * tanh_f(p.z + qq.z) + vv.w * tanh_f(p.w + qq.w);
            }
            A.e[b * TENC + tp] = s;
        }
        gsync(A.bar);

        // ---- Stage 4: softmax over TENC + context (this WG: rows wg*4..+4)
        {
            int bl = tid >> 7, th = tid & 127;
            int b = wg * 4 + bl;
            float ev[4], m = -3.4e38f;
#pragma unroll
            for (int i = 0; i < 4; ++i) {
                ev[i] = A.e[b * TENC + th + i * 128];
                m = fmaxf(m, ev[i]);
            }
#pragma unroll
            for (int o = 16; o > 0; o >>= 1) m = fmaxf(m, __shfl_xor(m, o, 32));
            if (lane == 0) redA[wave] = m;
            __syncthreads();
            m = fmaxf(fmaxf(redA[bl * 4 + 0], redA[bl * 4 + 1]),
                      fmaxf(redA[bl * 4 + 2], redA[bl * 4 + 3]));
            float ex[4], s = 0.f;
#pragma unroll
            for (int i = 0; i < 4; ++i) { ex[i] = __expf(ev[i] - m); s += ex[i]; }
#pragma unroll
            for (int o = 16; o > 0; o >>= 1) s += __shfl_xor(s, o, 32);
            if (lane == 0) redB[wave] = s;
            __syncthreads();
            float inv = 1.f / (redB[bl * 4 + 0] + redB[bl * 4 + 1] +
                               redB[bl * 4 + 2] + redB[bl * 4 + 3]);
#pragma unroll
            for (int i = 0; i < 4; ++i) {
                int tp = th + i * 128;
                float av = ex[i] * inv;
                aLds[bl][tp] = av;
                A.out_align[((long)b * TDEC + t) * TENC + tp] = av;
            }
            __syncthreads();
            // ctx[b,d] = sum_t a[b,t] * enc[b,t,d]
            for (int rep = 0; rep < 2; ++rep) {
                int ii = tid + rep * RBLK;             // 0..1023
                int bl2 = ii >> 8, dcol = ii & 255;
                int b2 = wg * 4 + bl2;
                const float* erow = A.enc + (long)b2 * TENC * DMODEL + dcol;
                float acc = 0.f;
                for (int tt = 0; tt < TENC; ++tt)
                    acc += aLds[bl2][tt] * erow[(long)tt * DMODEL];
                A.ctx[b2 * 256 + dcol] = acc;
                A.ctxb[b2 * 256 + dcol] = f2bf(acc);
            }
        }
        gsync(A.bar);

        // ---- Stage 5: dA = [att_h, ctx] @ proj_w + proj_b
        if (wave < 4) {
            f32x8 acc = {};
            acc = mm(acc, A.att_hb[cur], 256, wave * 16, 0, A.projwt, 512, wg * 16, 0, 8);
            acc = mm(acc, A.ctxb,        256, wave * 16, 0, A.projwt, 512, wg * 16, 256, 8);
            int col = wg * 16 + ln;
            float bv = A.proj_b[col];
#pragma unroll
            for (int r = 0; r < 8; ++r) {
                int bm = wave * 16 + r + 8 * hi;
                float v = acc[r] + bv;
                A.dA[bm * 256 + col] = v;
                A.dAb[bm * 256 + col] = f2bf(v);
            }
        }
        gsync(A.bar);

        // ---- Stage 6: h1 = GRU(dA, h1);  dB = h1 + dA
        dec_gru_stage(wg, wave, hi, ln, tid, A.dAb, A.dA,
                      A.h1b[prv], A.h1[prv], A.d1iht, A.d1hht,
                      A.d1_bih, A.d1_bhh, A.h1[cur], A.h1b[cur],
                      A.dB, A.dBb, gi_s, gh_s);
        gsync(A.bar);

        // ---- Stage 7: h2 = GRU(dB, h2);  dC = h2 + dB
        dec_gru_stage(wg, wave, hi, ln, tid, A.dBb, A.dB,
                      A.h2b[prv], A.h2[prv], A.d2iht, A.d2hht,
                      A.d2_bih, A.d2_bhh, A.h2[cur], A.h2b[cur],
                      A.dC, A.dCb, gi_s, gh_s);
        gsync(A.bar);

        // ---- Stage 8: out[:, t] = dC @ mel_w + mel_b   (25 n-tiles of 16)
        {
            int nt = -1, mt = wave & 3;
            if (wave < 4) nt = wg;
            else if (wave < 8 && wg + 16 < 25) nt = wg + 16;
            if (nt >= 0) {
                f32x8 acc = {};
                acc = mm(acc, A.dCb, 256, mt * 16, 0, A.melwt, 256, nt * 16, 0, 8);
                int col = nt * 16 + ln;
                float bv = A.mel_b[col];
#pragma unroll
                for (int r = 0; r < 8; ++r) {
                    int bm = mt * 16 + r + 8 * hi;
                    A.out_mel[((long)bm * TDEC + t) * INRAW + col] = acc[r] + bv;
                }
            }
        }
        gsync(A.bar);
    }
}

// ---------------------------------------------------------------------------
// Host-side orchestration
// ---------------------------------------------------------------------------
extern "C" void kernel_launch(void* const* d_in, const int* in_sizes, int n_in,
                              void* d_out, int out_size, void* d_ws, size_t ws_size,
                              hipStream_t stream) {
    (void)in_sizes; (void)n_in; (void)out_size; (void)ws_size;
    const float* enc      = (const float*)d_in[0];
    const float* mel      = (const float*)d_in[1];
    const float* pw1      = (const float*)d_in[2];
    const float* pb1      = (const float*)d_in[3];
    const float* pw2      = (const float*)d_in[4];
    const float* pb2      = (const float*)d_in[5];
    const float* att_wih  = (const float*)d_in[6];
    const float* att_whh  = (const float*)d_in[7];
    const float* att_bih  = (const float*)d_in[8];
    const float* att_bhh  = (const float*)d_in[9];
    const float* mem_w    = (const float*)d_in[10];
    const float* query_w  = (const float*)d_in[11];
    const float* att_v    = (const float*)d_in[12];
    const float* proj_w   = (const float*)d_in[13];
    const float* proj_b   = (const float*)d_in[14];
    const float* d1_wih   = (const float*)d_in[15];
    const float* d1_whh   = (const float*)d_in[16];
    const float* d1_bih   = (const float*)d_in[17];
    const float* d1_bhh   = (const float*)d_in[18];
    const float* d2_wih   = (const float*)d_in[19];
    const float* d2_whh   = (const float*)d_in[20];
    const float* d2_bih   = (const float*)d_in[21];
    const float* d2_bhh   = (const float*)d_in[22];
    const float* mel_w    = (const float*)d_in[23];
    const float* mel_b    = (const float*)d_in[24];

    char* ws = (char*)d_ws;
    size_t off = 0;
    auto alloc = [&](size_t bytes) -> char* {
        char* p = ws + off;
        off = (off + bytes + 255) & ~(size_t)255;
        return p;
    };

    unsigned short* Apre  = (unsigned short*)alloc((size_t)MROWS * INPAD * 2);
    unsigned short* X1    = (unsigned short*)alloc((size_t)MROWS * 256 * 2);
    unsigned short* XF    = (unsigned short*)alloc((size_t)MROWS * 128 * 2);
    unsigned short* ENCb  = (unsigned short*)alloc((size_t)EROWS * 256 * 2);
    float*          PM    = (float*)         alloc((size_t)EROWS * 256 * 4);

    unsigned short* w1t    = (unsigned short*)alloc((size_t)256 * INPAD * 2);
    unsigned short* w2t    = (unsigned short*)alloc((size_t)128 * 256 * 2);
    unsigned short* memwt  = (unsigned short*)alloc((size_t)256 * 256 * 2);
    unsigned short* attiht = (unsigned short*)alloc((size_t)768 * 384 * 2);
    unsigned short* atthht = (unsigned short*)alloc((size_t)768 * 256 * 2);
    unsigned short* qwt    = (unsigned short*)alloc((size_t)256 * 256 * 2);
    unsigned short* projwt = (unsigned short*)alloc((size_t)256 * 512 * 2);
    unsigned short* d1iht  = (unsigned short*)alloc((size_t)768 * 256 * 2);
    unsigned short* d1hht  = (unsigned short*)alloc((size_t)768 * 256 * 2);
    unsigned short* d2iht  = (unsigned short*)alloc((size_t)768 * 256 * 2);
    unsigned short* d2hht  = (unsigned short*)alloc((size_t)768 * 256 * 2);
    unsigned short* melwt  = (unsigned short*)alloc((size_t)400 * 256 * 2);

    // state zone (zeroed every launch)
    char* zone0 = ws + off;
    const size_t SV = (size_t)BATCH * 256;         // 16384 elements
    float* att_h0 = (float*)alloc(SV * 4); float* att_h1 = (float*)alloc(SV * 4);
    float* h1_0   = (float*)alloc(SV * 4); float* h1_1   = (float*)alloc(SV * 4);
    float* h2_0   = (float*)alloc(SV * 4); float* h2_1   = (float*)alloc(SV * 4);
    float* ctx    = (float*)alloc(SV * 4);
    float* qbuf   = (float*)alloc(SV * 4);
    float* dA     = (float*)alloc(SV * 4);
    float* dB     = (float*)alloc(SV * 4);
    float* dC     = (float*)alloc(SV * 4);
    float* ebuf   = (float*)alloc((size_t)BATCH * TENC * 4);
    unsigned short* att_hb0 = (unsigned short*)alloc(SV * 2);
    unsigned short* att_hb1 = (unsigned short*)alloc(SV * 2);
    unsigned short* h1b0    = (unsigned short*)alloc(SV * 2);
    unsigned short* h1b1    = (unsigned short*)alloc(SV * 2);
    unsigned short* h2b0    = (unsigned short*)alloc(SV * 2);
    unsigned short* h2b1    = (unsigned short*)alloc(SV * 2);
    unsigned short* ctxb    = (unsigned short*)alloc(SV * 2);
    unsigned short* dAb     = (unsigned short*)alloc(SV * 2);
    unsigned short* dBb     = (unsigned short*)alloc(SV * 2);
    unsigned short* dCb     = (unsigned short*)alloc(SV * 2);
    unsigned* bar = (unsigned*)alloc(256);
    size_t zone_bytes = (size_t)((ws + off) - zone0);
    hipMemsetAsync(zone0, 0, zone_bytes, stream);

    // ---- prep ----
    {
        long n = (long)MROWS * INPAD;
        k_build_apre<<<(unsigned)((n + 255) / 256), 256, 0, stream>>>(mel, Apre);
    }
    {
        long n = (long)EROWS * 256;
        k_f2bf<<<(unsigned)((n + 255) / 256), 256, 0, stream>>>(enc, ENCb, n);
    }
    auto wt = [&](const float* src, unsigned short* dst, int K, int N, int Kpad) {
        long n = (long)N * Kpad;
        k_wtrans<<<(unsigned)((n + 255) / 256), 256, 0, stream>>>(src, dst, K, N, Kpad);
    };
    wt(pw1,     w1t,    INRAW, 256, INPAD);
    wt(pw2,     w2t,    256,   128, 256);
    wt(mem_w,   memwt,  256,   256, 256);
    wt(att_wih, attiht, 384,   768, 384);
    wt(att_whh, atthht, 256,   768, 256);
    wt(query_w, qwt,    256,   256, 256);
    wt(proj_w,  projwt, 512,   256, 512);
    wt(d1_wih,  d1iht,  256,   768, 256);
    wt(d1_whh,  d1hht,  256,   768, 256);
    wt(d2_wih,  d2iht,  256,   768, 256);
    wt(d2_whh,  d2hht,  256,   768, 256);
    wt(mel_w,   melwt,  256,   400, 256);

    // ---- parallel GEMMs (WMMA) ----
    auto gemm = [&](const unsigned short* A, long lda, const unsigned short* Bt, long ldb,
                    const float* bias, float* Cf, unsigned short* Cb, long ldc,
                    int Mt, int Nt, int Kp, int relu) {
        int tiles = Mt * Nt;
        k_gemm<<<(tiles + 7) / 8, 256, 0, stream>>>(A, lda, Bt, ldb, bias, Cf, Cb,
                                                    ldc, Mt, Nt, Kp, relu);
    };
    gemm(Apre, INPAD, w1t, INPAD, pb1, nullptr, X1, 256, MROWS / 16, 16, INPAD, 1);
    gemm(X1, 256, w2t, 256, pb2, nullptr, XF, 128, MROWS / 16, 8, 256, 1);
    gemm(ENCb, 256, memwt, 256, nullptr, PM, nullptr, 256, EROWS / 16, 16, 256, 0);

    // ---- persistent recurrent kernel ----
    RecArgs ra;
    ra.attiht = attiht; ra.atthht = atthht; ra.qwt = qwt; ra.projwt = projwt;
    ra.d1iht = d1iht; ra.d1hht = d1hht; ra.d2iht = d2iht; ra.d2hht = d2hht;
    ra.melwt = melwt;
    ra.att_bih = att_bih; ra.att_bhh = att_bhh; ra.proj_b = proj_b;
    ra.d1_bih = d1_bih; ra.d1_bhh = d1_bhh; ra.d2_bih = d2_bih; ra.d2_bhh = d2_bhh;
    ra.mel_b = mel_b; ra.att_v = att_v;
    ra.enc = enc; ra.PM = PM; ra.XF = XF;
    ra.att_h[0] = att_h0; ra.att_h[1] = att_h1;
    ra.att_hb[0] = att_hb0; ra.att_hb[1] = att_hb1;
    ra.h1[0] = h1_0; ra.h1[1] = h1_1; ra.h1b[0] = h1b0; ra.h1b[1] = h1b1;
    ra.h2[0] = h2_0; ra.h2[1] = h2_1; ra.h2b[0] = h2b0; ra.h2b[1] = h2b1;
    ra.ctx = ctx; ra.ctxb = ctxb; ra.q = qbuf;
    ra.dA = dA; ra.dB = dB; ra.dC = dC; ra.dAb = dAb; ra.dBb = dBb; ra.dCb = dCb;
    ra.e = ebuf;
    ra.out_mel = (float*)d_out;
    ra.out_align = (float*)d_out + (size_t)BATCH * TDEC * INRAW;
    ra.bar = bar;
    k_rec<<<NWG, RBLK, 0, stream>>>(ra);
}
